// LMNet_14336600834411
// MI455X (gfx1250) — matrix operand order repaired and checked by reading, not compiled, not run on hardware
//
#include <hip/hip_runtime.h>
#include <hip/hip_bf16.h>

// ---------------------------------------------------------------------------
// Problem constants (match reference)
// ---------------------------------------------------------------------------
#define BB     2
#define DD     128
#define HH     128
#define WW     128
#define HWQ    (HH*WW)                 // 16384
#define NVOX   (DD*HH*WW)              // 2,097,152
#define NTOT   (BB*NVOX)               // 4,194,304
#define EE     500000
#define KK     32
#define NFEAT  16
#define NBLK   3

typedef _Float16 v8h  __attribute__((ext_vector_type(8)));
typedef _Float16 v16h __attribute__((ext_vector_type(16)));
typedef float    v8f  __attribute__((ext_vector_type(8)));

// ---------------------------------------------------------------------------
// Workspace layout (bytes, 256-aligned chunks)
// ---------------------------------------------------------------------------
#define OFF_SCALES   ((size_t)0)                         // 4 floats: scale[2], invscale[2]
#define OFF_PART     ((size_t)256)                       // 2048 floats partial sums
#define OFF_WPREP    (OFF_PART + 8448)                   // 3 * 22016 halves (f16 swizzled weights)
#define WPREP_HALVES ((size_t)22016)                     // per block: 512 + 3*7168
#define OFF_BP       (OFF_WPREP + 132096 + 256)          // NTOT f32
#define OFF_XU       (OFF_BP + (size_t)NTOT*4)           // NTOT f32
#define OFF_XC       (OFF_XU + (size_t)NTOT*4)           // NTOT f32
#define OFF_XIN      (OFF_XC + (size_t)NTOT*4)           // NTOT f16
#define OFF_CV       (OFF_XIN + (size_t)NTOT*2)          // NTOT f32
#define OFF_BUFA     (OFF_CV + (size_t)NTOT*4)           // NTOT*16 f16 channels-last
#define OFF_BUFB     (OFF_BUFA + (size_t)NTOT*16*2)      // NTOT*16 f16 channels-last

// Async 16B global -> LDS copy (per-lane, tracked by ASYNCcnt).
// ldsoff = wave-relative LDS byte offset (low 32 bits of generic LDS pointer),
// gaddr  = 64-bit global byte address.
__device__ __forceinline__ void async_g2l_b128(unsigned ldsoff, unsigned long long gaddr) {
  asm volatile("global_load_async_to_lds_b128 %0, %1, off"
               :
               : "v"(ldsoff), "v"(gaddr)
               : "memory");
}
__device__ __forceinline__ void wait_async0() {
  asm volatile("s_wait_asynccnt 0x0" ::: "memory");
}

// ---------------------------------------------------------------------------
// Deterministic per-sample mean (two-stage tree reduction, no float atomics)
// ---------------------------------------------------------------------------
__global__ __launch_bounds__(256) void reduce1_kernel(const float* __restrict__ x,
                                                      float* __restrict__ partials) {
  __shared__ float red[256];
  const int b = blockIdx.y;
  const size_t base = (size_t)b*NVOX + (size_t)blockIdx.x*2048 + (size_t)threadIdx.x*8;
  float s = 0.f;
#pragma unroll
  for (int j = 0; j < 8; ++j) s += x[base + j];
  red[threadIdx.x] = s;
  __syncthreads();
  for (int off = 128; off > 0; off >>= 1) {
    if (threadIdx.x < off) red[threadIdx.x] += red[threadIdx.x + off];
    __syncthreads();
  }
  if (threadIdx.x == 0) partials[(size_t)b*1024 + blockIdx.x] = red[0];
}

__global__ __launch_bounds__(256) void reduce2_kernel(const float* __restrict__ partials,
                                                      float* __restrict__ scales) {
  __shared__ float red[256];
  for (int b = 0; b < BB; ++b) {
    float s = partials[(size_t)b*1024 + threadIdx.x]
            + partials[(size_t)b*1024 + threadIdx.x + 256]
            + partials[(size_t)b*1024 + threadIdx.x + 512]
            + partials[(size_t)b*1024 + threadIdx.x + 768];
    red[threadIdx.x] = s;
    __syncthreads();
    for (int off = 128; off > 0; off >>= 1) {
      if (threadIdx.x < off) red[threadIdx.x] += red[threadIdx.x + off];
      __syncthreads();
    }
    if (threadIdx.x == 0) {
      float mean = red[0] / (float)NVOX;
      scales[b]     = mean;
      scales[2 + b] = 1.0f / mean;
    }
    __syncthreads();
  }
}

// ---------------------------------------------------------------------------
// Zero a float buffer (float4 stores)
// ---------------------------------------------------------------------------
__global__ __launch_bounds__(256) void zero4_kernel(float4* __restrict__ p, int n4) {
  int i = blockIdx.x * 256 + threadIdx.x;
  if (i < n4) { float4 z; z.x = z.y = z.z = z.w = 0.f; p[i] = z; }
}

// ---------------------------------------------------------------------------
// Weight pre-swizzle to f16:
//   wh1s [2][16][16]  : per-lane-half swizzled B fragment for the 1->16 layer
//   wh2/3/4 [28][16][16] (tap, co, ci), tap 27 = zero pad (for K=32 pairing)
// ---------------------------------------------------------------------------
__global__ __launch_bounds__(256) void prep_weights_kernel(
    const float* __restrict__ w1,   // [16][1][27]
    const float* __restrict__ w2,   // [16][16][27]
    const float* __restrict__ w3,   // [16][16][27]
    const float* __restrict__ w4,   // [1][16][27]
    _Float16* __restrict__ dst) {   // 22016 halves
  int g = blockIdx.x * 256 + threadIdx.x;
  if (g >= 22016) return;
  if (g < 512) {
    int h = g >> 8, rem = g & 255, n = rem >> 4, j = rem & 15;
    int Kk = (h == 0) ? ((j < 8) ? j : j + 8) : ((j < 8) ? j + 8 : j + 16);
    dst[g] = (Kk < 27) ? (_Float16)w1[n*27 + Kk] : (_Float16)0.f;
  } else {
    int q = g - 512;
    int sel = q / 7168, r2 = q % 7168;
    int tap = r2 >> 8, co = (r2 >> 4) & 15, ci = r2 & 15;
    float v = 0.f;
    if (tap < 27) {
      if (sel == 0)      v = w2[(co*16 + ci)*27 + tap];
      else if (sel == 1) v = w3[(co*16 + ci)*27 + tap];
      else               v = (co == 0) ? w4[ci*27 + tap] : 0.f;
    }
    dst[g] = (_Float16)v;
  }
}

// ---------------------------------------------------------------------------
// List-mode forward projection + scatter back-projection (1 thread / event)
// ---------------------------------------------------------------------------
__global__ __launch_bounds__(256) void lm_fwd_scatter_kernel(
    const float* __restrict__ xc, const int* __restrict__ idx,
    const float* __restrict__ w, const float* __restrict__ contam,
    float* __restrict__ bp) {
  size_t eg = (size_t)blockIdx.x * 256 + threadIdx.x;
  if (eg >= (size_t)BB * EE) return;
  const int b = (int)(eg / EE);
  const int4*   ip = (const int4*)(idx + eg * KK);
  const float4* wp = (const float4*)(w + eg * KK);
  const float* xb = xc + (size_t)b * NVOX;
  int4 iv[8]; float4 wv[8];
#pragma unroll
  for (int j = 0; j < 8; ++j) { iv[j] = ip[j]; wv[j] = wp[j]; }
  float fwd = contam[eg];
#pragma unroll
  for (int j = 0; j < 8; ++j) {
    fwd += wv[j].x * xb[iv[j].x];
    fwd += wv[j].y * xb[iv[j].y];
    fwd += wv[j].z * xb[iv[j].z];
    fwd += wv[j].w * xb[iv[j].w];
  }
  const float inv = 1.0f / fwd;
  float* bb = bp + (size_t)b * NVOX;
#pragma unroll
  for (int j = 0; j < 8; ++j) {
    atomicAdd(&bb[iv[j].x], wv[j].x * inv);
    atomicAdd(&bb[iv[j].y], wv[j].y * inv);
    atomicAdd(&bb[iv[j].z], wv[j].z * inv);
    atomicAdd(&bb[iv[j].w], wv[j].w * inv);
  }
}

// ---------------------------------------------------------------------------
// xu = x - precond*(adj - bp);  xin_f16 = xu * invscale
// ---------------------------------------------------------------------------
__global__ __launch_bounds__(256) void update1_kernel(
    const float* __restrict__ xc, const float* __restrict__ pre,
    const float* __restrict__ adj, const float* __restrict__ bp,
    const float* __restrict__ scales,
    float* __restrict__ xu, _Float16* __restrict__ xin) {
  size_t i = (size_t)blockIdx.x * 256 + threadIdx.x;
  if (i >= (size_t)NTOT) return;
  const int b = (int)(i / NVOX);
  float v = xc[i] - pre[i] * (adj[i] - bp[i]);
  xu[i] = v;
  xin[i] = (_Float16)(v * scales[2 + b]);
}

// ---------------------------------------------------------------------------
// x_next = relu(xu - convout * scale)
// ---------------------------------------------------------------------------
__global__ __launch_bounds__(256) void update2_kernel(
    const float* __restrict__ xu, const float* __restrict__ cv,
    const float* __restrict__ scales, float* __restrict__ out) {
  size_t i = (size_t)blockIdx.x * 256 + threadIdx.x;
  if (i >= (size_t)NTOT) return;
  const int b = (int)(i / NVOX);
  out[i] = fmaxf(0.f, xu[i] - cv[i] * scales[b]);
}

// ---------------------------------------------------------------------------
// Conv layer 1: 1 -> 16 channels, one WMMA (K=27 padded to 32) per 16-voxel strip
// in:  xin [B][nvox] f16         out: [B][nvox][16] f16 channels-last (+bias)
// ---------------------------------------------------------------------------
__global__ __launch_bounds__(256) void conv1_kernel(
    const _Float16* __restrict__ xin, const _Float16* __restrict__ wh1s, // [2][16][16]
    const float* __restrict__ bias, _Float16* __restrict__ out16) {
  const int tid = threadIdx.x;
  const int b = blockIdx.y;
  const int zy = blockIdx.x;
  const int z = zy >> 7, y = zy & 127;

  __shared__ alignas(32) _Float16 tile1[9 * 130];
  __shared__ alignas(32) _Float16 w1s[512];

  for (int s = tid; s < 9 * 130; s += 256) {
    const int r = s / 130, xs = s % 130;
    const int zz = z + r/3 - 1, yy = y + r%3 - 1, xi = xs - 1;
    _Float16 v = (_Float16)0.f;
    if ((unsigned)zz < 128u && (unsigned)yy < 128u && (unsigned)xi < 128u)
      v = xin[(size_t)b*NVOX + (size_t)zz*HWQ + yy*WW + xi];
    tile1[s] = v;
  }
  {
    const uint4* gw = (const uint4*)wh1s;
    uint4* sw = (uint4*)w1s;
    if (tid < 64) sw[tid] = gw[tid];
  }
  __syncthreads();

  const int lane = tid & 31, wv = tid >> 5;
  const int m = lane & 15, hid = lane >> 4;
  const int x0 = wv * 16 + m;

  v16h av;
#pragma unroll
  for (int j = 0; j < 16; ++j) {
    int Kk = (hid == 0) ? ((j < 8) ? j : j + 8) : ((j < 8) ? j + 8 : j + 16);
    int t = (Kk < 27) ? Kk : 26;                     // K>=27 -> B is zero, any valid addr
    int row = (t/9)*3 + (t%9)/3, dx = t%3;
    av[j] = tile1[row*130 + x0 + dx];
  }
  const v16h bv = *(const v16h*)&w1s[hid*256 + m*16];
  v8f c = {0.f,0.f,0.f,0.f,0.f,0.f,0.f,0.f};
  c = __builtin_amdgcn_wmma_f32_16x16x32_f16(false, av, false, bv, (short)0, c, false, false);

  const float bn = bias[m];
  const size_t rowbase = (size_t)b*NVOX + (size_t)z*HWQ + (size_t)y*WW;
#pragma unroll
  for (int r = 0; r < 8; ++r) {
    const int xo = wv*16 + hid*8 + r;
    out16[((rowbase + xo) << 4) + m] = (_Float16)(c[r] + bn);
  }
}

// ---------------------------------------------------------------------------
// Conv 16 -> 16 (mode 0: bias; mode 1: bias+PReLU; mode 2: 16->1, f32 out)
// 14 WMMAs per 16-voxel strip (two 3x3x3 taps packed per K=32 step).
// Tile + weight staging uses async global->LDS (ASYNCcnt) instead of VGPR
// round-trips; only zero-filled halo slots use plain ds stores.
// ---------------------------------------------------------------------------
__global__ __launch_bounds__(256) void conv16_kernel(
    const _Float16* __restrict__ in16,   // [B][nvox][16] f16 channels-last
    const _Float16* __restrict__ wh,     // [28][16][16] (tap, co, ci) f16
    const float* __restrict__ bias,
    const float* __restrict__ alpha,     // mode 1 only
    _Float16* __restrict__ out16,        // modes 0/1
    float* __restrict__ out1,            // mode 2
    int mode) {
  const int tid = threadIdx.x;
  const int b = blockIdx.y;
  const int zy = blockIdx.x;
  const int z = zy >> 7, y = zy & 127;

  __shared__ alignas(32) _Float16 tile[9 * 130 * 16];   // 37,440 B
  __shared__ alignas(32) _Float16 wlds[28 * 16 * 16];   // 14,336 B

  // stage halo tile; channels-last rows are contiguous 32B (= 2x16B) chunks
  for (int s = tid; s < 9 * 130; s += 256) {
    const int r = s / 130, xs = s % 130;
    const int zz = z + r/3 - 1, yy = y + r%3 - 1, xi = xs - 1;
    _Float16* sp = &tile[s * 16];
    if ((unsigned)zz < 128u && (unsigned)yy < 128u && (unsigned)xi < 128u) {
      const _Float16* gp = in16 + (((size_t)b*NVOX + (size_t)zz*HWQ + (size_t)yy*WW + xi) << 4);
      const unsigned loff = (unsigned)(size_t)sp;
      const unsigned long long ga = (unsigned long long)(size_t)gp;
      async_g2l_b128(loff,      ga);
      async_g2l_b128(loff + 16, ga + 16);
    } else {
      uint4 zz4; zz4.x = zz4.y = zz4.z = zz4.w = 0u;
      ((uint4*)sp)[0] = zz4;
      ((uint4*)sp)[1] = zz4;
    }
  }
  // stage weights (28*16*16 halves = 896 x 16B)
  for (int t2 = tid; t2 < 896; t2 += 256) {
    async_g2l_b128((unsigned)(size_t)&wlds[t2 * 8],
                   (unsigned long long)(size_t)(wh + t2 * 8));
  }
  wait_async0();
  __syncthreads();

  const int lane = tid & 31, wv = tid >> 5;
  const int m = lane & 15, hid = lane >> 4;
  const int cib = hid * 8;
  const int x0 = wv * 16 + m;

  v8f c = {0.f,0.f,0.f,0.f,0.f,0.f,0.f,0.f};
#pragma unroll
  for (int p = 0; p < 14; ++p) {
    const int t0 = 2*p;
    const int t1 = 2*p + 1;
    const int t1a = (t1 < 27) ? t1 : 26;            // tap 27 has zero B weights
    const int r0 = (t0/9)*3 + (t0%9)/3,  dx0 = t0 % 3;
    const int r1 = (t1a/9)*3 + (t1a%9)/3, dx1 = t1a % 3;
    // A fragment: lanes 0-15 hold K 0-7 / 16-23, lanes 16-31 hold K 8-15 / 24-31
    v8h alo = *(const v8h*)&tile[(r0*130 + x0 + dx0)*16 + cib];
    v8h ahi = *(const v8h*)&tile[(r1*130 + x0 + dx1)*16 + cib];
    v16h av = __builtin_shufflevector(alo, ahi, 0,1,2,3,4,5,6,7,8,9,10,11,12,13,14,15);
    // B fragment: same K order as A (elements 0-7 tap t0 ci cib.., 8-15 tap t1 ci cib..)
    v8h blo = *(const v8h*)&wlds[t0*256 + m*16 + cib];
    v8h bhi = *(const v8h*)&wlds[t1*256 + m*16 + cib];
    v16h bv = __builtin_shufflevector(blo, bhi, 0,1,2,3,4,5,6,7,8,9,10,11,12,13,14,15);
    c = __builtin_amdgcn_wmma_f32_16x16x32_f16(false, av, false, bv, (short)0, c, false, false);
  }

  const size_t rowbase = (size_t)b*NVOX + (size_t)z*HWQ + (size_t)y*WW;
  if (mode == 2) {
    if (m == 0) {
      const float b0 = bias[0];
#pragma unroll
      for (int r = 0; r < 8; ++r) {
        const int xo = wv*16 + hid*8 + r;
        out1[rowbase + xo] = c[r] + b0;
      }
    }
  } else {
    const float bn = bias[m];
    const float al = (mode == 1) ? alpha[0] : 0.f;
#pragma unroll
    for (int r = 0; r < 8; ++r) {
      const int xo = wv*16 + hid*8 + r;
      float v = c[r] + bn;
      if (mode == 1) v = (v > 0.f) ? v : al * v;
      out16[((rowbase + xo) << 4) + m] = (_Float16)v;
    }
  }
}

// ---------------------------------------------------------------------------
// Launch sequence
// ---------------------------------------------------------------------------
extern "C" void kernel_launch(void* const* d_in, const int* in_sizes, int n_in,
                              void* d_out, int out_size, void* d_ws, size_t ws_size,
                              hipStream_t stream) {
  (void)in_sizes; (void)n_in; (void)out_size; (void)ws_size;

  const float* x0   = (const float*)d_in[0];
  const int*   lmi  = (const int*)d_in[1];
  const float* lmw  = (const float*)d_in[2];
  const float* con  = (const float*)d_in[3];
  const float* adj  = (const float*)d_in[4];
  const float* pre  = (const float*)d_in[5];

  char* ws = (char*)d_ws;
  float*     scales = (float*)(ws + OFF_SCALES);
  float*     part   = (float*)(ws + OFF_PART);
  _Float16*  wprep  = (_Float16*)(ws + OFF_WPREP);
  float*     bp     = (float*)(ws + OFF_BP);
  float*     XU     = (float*)(ws + OFF_XU);
  float*     XC     = (float*)(ws + OFF_XC);
  _Float16*  XIN    = (_Float16*)(ws + OFF_XIN);
  float*     CV     = (float*)(ws + OFF_CV);
  _Float16*  BUFA   = (_Float16*)(ws + OFF_BUFA);
  _Float16*  BUFB   = (_Float16*)(ws + OFF_BUFB);

  // sample_scales from the ORIGINAL x, once
  reduce1_kernel<<<dim3(1024, BB), 256, 0, stream>>>(x0, part);
  reduce2_kernel<<<1, 256, 0, stream>>>(part, scales);

  // pre-swizzle conv weights to f16 fragments
  for (int i = 0; i < NBLK; ++i) {
    const int base = 6 + i*9;
    prep_weights_kernel<<<86, 256, 0, stream>>>(
        (const float*)d_in[base + 0],   // w1
        (const float*)d_in[base + 2],   // w2
        (const float*)d_in[base + 5],   // w3
        (const float*)d_in[base + 7],   // w4
        wprep + (size_t)i * WPREP_HALVES);
  }

  const int nEl   = NTOT;
  const int gElem = (nEl + 255) / 256;          // 16384
  const int gEv   = (BB*EE + 255) / 256;
  const dim3 gConv(HWQ, BB);

  const float* xc = x0;
  for (int i = 0; i < NBLK; ++i) {
    const int base = 6 + i*9;
    const float* b1 = (const float*)d_in[base + 1];
    const float* b2 = (const float*)d_in[base + 3];
    const float* al = (const float*)d_in[base + 4];
    const float* b3 = (const float*)d_in[base + 6];
    const float* b4 = (const float*)d_in[base + 8];
    _Float16* wh1s = wprep + (size_t)i * WPREP_HALVES;
    _Float16* wh2  = wh1s + 512;
    _Float16* wh3  = wh2 + 7168;
    _Float16* wh4  = wh3 + 7168;

    zero4_kernel<<<4096, 256, 0, stream>>>((float4*)bp, nEl / 4);
    lm_fwd_scatter_kernel<<<gEv, 256, 0, stream>>>(xc, lmi, lmw, con, bp);
    update1_kernel<<<gElem, 256, 0, stream>>>(xc, pre, adj, bp, scales, XU, XIN);

    conv1_kernel<<<gConv, 256, 0, stream>>>(XIN, wh1s, b1, BUFA);
    conv16_kernel<<<gConv, 256, 0, stream>>>(BUFA, wh2, b2, al, BUFB, nullptr, 1);
    conv16_kernel<<<gConv, 256, 0, stream>>>(BUFB, wh3, b3, nullptr, BUFA, nullptr, 0);
    conv16_kernel<<<gConv, 256, 0, stream>>>(BUFA, wh4, b4, nullptr, nullptr, CV, 2);

    float* xn = (i == NBLK - 1) ? (float*)d_out : XC;
    update2_kernel<<<gElem, 256, 0, stream>>>(XU, CV, scales, xn);
    xc = xn;
  }
}
